// MultiheadAttention_5626407158199
// MI455X (gfx1250) — compile-verified, matrix operand order checked by reading, not run on hardware
//
#include <hip/hip_runtime.h>

// ---------------------------------------------------------------------------
// MI455X (gfx1250) multi-head attention, bf16 WMMA + f32 accumulate.
// Pipeline: cvt(f32->bf16) -> 3x GEMM (proj, bf16 out) -> flash-attn -> GEMM (Wo, f32 out)
// Data movement: double-buffered global_load_async_to_lds_b128 (ASYNCcnt).
// Softmax: softcap bounds |logits| < 50 -> fixed zero offset, no online rescale.
// Fragment loads are batched ahead of each WMMA chain so LDS latency overlaps
// matrix ops instead of forcing s_wait_dscnt 0 per WMMA.
// ---------------------------------------------------------------------------

typedef __attribute__((ext_vector_type(16))) __bf16 v16bf;
typedef __attribute__((ext_vector_type(8)))  __bf16 v8bf;
typedef __attribute__((ext_vector_type(4)))  __bf16 v4bf;
typedef __attribute__((ext_vector_type(8)))  float  v8f;

#define NUM_HEADS 16
#define ALPHA     1.0f
#define SOFTCAP   50.0f

// Low 32 bits of a flat shared-aperture address == LDS byte offset (ISA 10.2).
static __device__ __forceinline__ unsigned lds_off_u32(const void* p) {
  return (unsigned)(uintptr_t)p;
}

// Async global->LDS copy, GVS mode: saddr(uniform base) + 32-bit vgpr offset.
// Tracked by ASYNCcnt; data lands in LDS without touching VGPRs.
static __device__ __forceinline__ void async_copy_b128(unsigned lds_dst,
                                                       const void* sbase,
                                                       unsigned byte_off) {
  asm volatile("global_load_async_to_lds_b128 %0, %1, %2"
               :: "v"(lds_dst), "v"(byte_off), "s"(sbase)
               : "memory");
}
static __device__ __forceinline__ void wait_async0() {
  asm volatile("s_wait_asynccnt 0x0" ::: "memory");
}

static __device__ __forceinline__ float fast_tanh(float x) {
#if __has_builtin(__builtin_amdgcn_tanhf)
  return __builtin_amdgcn_tanhf(x);   // gfx1250 v_tanh_f32 (TRANS)
#else
  return tanhf(x);
#endif
}

// CDNA5 16-bit A/B fragment from a row-major [row][k] LDS tile (ISA 7.12.2):
// lanes 0-15 hold K chunks {0..7,16..23}, lanes 16-31 hold {8..15,24..31};
// operand row (M for A, N for B) = lane & 15. Two aligned b128 loads.
static __device__ __forceinline__ v16bf ldsfrag(const __bf16* base, int ldst, int lane) {
  const __bf16* p = base + (lane & 15) * ldst + ((lane >> 4) << 3);
  v8bf lo = *(const v8bf*)(p);
  v8bf hi = *(const v8bf*)(p + 16);
  return __builtin_shufflevector(lo, hi, 0,1,2,3,4,5,6,7,8,9,10,11,12,13,14,15);
}

static __device__ __forceinline__ v8f wmma_bf16(v16bf a, v16bf b, v8f c) {
  return __builtin_amdgcn_wmma_f32_16x16x32_bf16(false, a, false, b, (short)0, c, false, false);
}

// ---------------------------------------------------------------------------
// f32 -> bf16 conversion (x4 vectorized; all counts are multiples of 4)
// ---------------------------------------------------------------------------
__global__ __launch_bounds__(256) void cvt_f32_bf16(const float* __restrict__ in,
                                                    __bf16* __restrict__ out, int n) {
  int i = (blockIdx.x * 256 + threadIdx.x) * 4;
  if (i + 3 < n) {
    float4 f = *(const float4*)(in + i);
    v4bf o = {(__bf16)f.x, (__bf16)f.y, (__bf16)f.z, (__bf16)f.w};
    *(v4bf*)(out + i) = o;
  }
}

// ---------------------------------------------------------------------------
// GEMM: OUT[M,N] = X[M,K] @ W[N,K]^T (bf16 in, f32 acc). Block tile 256x128,
// K-step 32, 8 waves (4M x 2N), wave tile 64x64 = 16 WMMA per K-step.
// Double-buffered async LDS staging; B fragments streamed in pairs.
// ---------------------------------------------------------------------------
template <bool OUT_BF16>
__global__ __launch_bounds__(256) void gemm_xwt_bf16(const __bf16* __restrict__ X,
                                                     const __bf16* __restrict__ W,
                                                     void* __restrict__ OUT,
                                                     int M, int N, int K) {
  constexpr int LDT = 40;  // bf16 row stride: 80B, 16B aligned, bank-skewed
  __shared__ __bf16 As[2][256 * LDT];
  __shared__ __bf16 Ws[2][128 * LDT];

  const int t = threadIdx.x;
  const int lane = t & 31;
  const int wid = t >> 5;
  const int m0 = blockIdx.y * 256;
  const int n0 = blockIdx.x * 128;
  const int waveM = (wid >> 1) * 64;
  const int waveN = (wid & 1) * 64;

  const __bf16* Xb = X + (size_t)m0 * K;
  const __bf16* Wb = W + (size_t)n0 * K;

  auto stage = [&](int buf, int k0) {
#pragma unroll
    for (int i = 0; i < 4; ++i) {  // A panel: 256x32 bf16 = 1024 x 16B
      int idx = t + i * 256;
      int r = idx >> 2, c = (idx & 3) << 3;
      async_copy_b128(lds_off_u32(&As[buf][r * LDT + c]), Xb,
                      (unsigned)((r * K + k0 + c) * 2));
    }
#pragma unroll
    for (int i = 0; i < 2; ++i) {  // W panel: 128x32 bf16 = 512 x 16B
      int idx = t + i * 256;
      int r = idx >> 2, c = (idx & 3) << 3;
      async_copy_b128(lds_off_u32(&Ws[buf][r * LDT + c]), Wb,
                      (unsigned)((r * K + k0 + c) * 2));
    }
  };

  v8f acc[4][4];
#pragma unroll
  for (int i = 0; i < 4; ++i)
#pragma unroll
    for (int j = 0; j < 4; ++j) acc[i][j] = (v8f)(0.0f);

  stage(0, 0);
  wait_async0();
  __syncthreads();

  const int nk = K / 32;
  for (int kt = 0; kt < nk; ++kt) {
    const int cur = kt & 1;
    if (kt + 1 < nk) stage(cur ^ 1, (kt + 1) * 32);  // overlap with WMMA below

    v16bf a[4];
#pragma unroll
    for (int mt = 0; mt < 4; ++mt)
      a[mt] = ldsfrag(&As[cur][(waveM + mt * 16) * LDT], LDT, lane);
#pragma unroll
    for (int np = 0; np < 2; ++np) {  // B fragments in pairs: 4 loads, 8 WMMA
      v16bf b0 = ldsfrag(&Ws[cur][(waveN + (np * 2 + 0) * 16) * LDT], LDT, lane);
      v16bf b1 = ldsfrag(&Ws[cur][(waveN + (np * 2 + 1) * 16) * LDT], LDT, lane);
#pragma unroll
      for (int mt = 0; mt < 4; ++mt) acc[mt][np * 2 + 0] = wmma_bf16(a[mt], b0, acc[mt][np * 2 + 0]);
#pragma unroll
      for (int mt = 0; mt < 4; ++mt) acc[mt][np * 2 + 1] = wmma_bf16(a[mt], b1, acc[mt][np * 2 + 1]);
    }

    if (kt + 1 < nk) wait_async0();
    __syncthreads();
  }

  // Epilogue: C layout -> lane holds column n = lane&15, rows (lane>>4)*8 + r
  const int nn = lane & 15;
  const int rbase = (lane >> 4) << 3;
#pragma unroll
  for (int mt = 0; mt < 4; ++mt) {
#pragma unroll
    for (int nt = 0; nt < 4; ++nt) {
      int gm = m0 + waveM + mt * 16 + rbase;
      int gn = n0 + waveN + nt * 16 + nn;
#pragma unroll
      for (int r = 0; r < 8; ++r) {
        float v = acc[mt][nt][r];
        if (OUT_BF16)
          ((__bf16*)OUT)[(size_t)(gm + r) * N + gn] = (__bf16)v;
        else
          ((float*)OUT)[(size_t)(gm + r) * N + gn] = v;
      }
    }
  }
}

// ---------------------------------------------------------------------------
// Flash attention, per (b,h) and q-tile of 128 (8 waves x 16 q-rows).
// Logits computed transposed (S^T = K * Q^T): per-query softmax lives in one
// lane column; the P^T tile in C-layout is bit-identical to the B-operand
// layout for ctx^T = V^T * P, so probs feed the second WMMA chain in-register.
// K tiles double-buffered via async-to-LDS; V transposed through VGPRs.
// All fragment loads for a WMMA chain are issued before the chain.
// ---------------------------------------------------------------------------
__global__ __launch_bounds__(256) void flash_attn(const __bf16* __restrict__ Qp,
                                                  const __bf16* __restrict__ Kp,
                                                  const __bf16* __restrict__ Vp,
                                                  __bf16* __restrict__ Ctx,
                                                  int B, int S, int E, int H) {
  constexpr int D = 128;
  constexpr int LQ = 136, LK = 136, LV = 40;
  __shared__ __bf16 Qs[128 * LQ];      // [q_local][d]
  __shared__ __bf16 Ks[2][32 * LK];    // [kv_local][d], double buffered
  __shared__ __bf16 Vt[2][128 * LV];   // [d][kv_local], double buffered

  const int t = threadIdx.x;
  const int lane = t & 31;
  const int wid = t >> 5;
  const int bh = blockIdx.y;
  const int b = bh / H, h = bh % H;
  const int q0 = blockIdx.x * 128;

  const __bf16* Qb = Qp + (size_t)b * S * E + (size_t)h * D;
  const __bf16* Kb = Kp + (size_t)b * S * E + (size_t)h * D;
  const __bf16* Vb = Vp + (size_t)b * S * E + (size_t)h * D;
  __bf16* Cb = Ctx + (size_t)b * S * E + (size_t)h * D;

  // Q tile (128x128) async -> LDS
#pragma unroll
  for (int i = 0; i < 8; ++i) {
    int idx = t + i * 256;
    int r = idx >> 4, c = (idx & 15) << 3;
    async_copy_b128(lds_off_u32(&Qs[r * LQ + c]), Qb,
                    (unsigned)(((q0 + r) * E + c) * 2));
  }

  auto stageK = [&](int buf, int kv0) {
#pragma unroll
    for (int i = 0; i < 2; ++i) {
      int idx = t + i * 256;
      int r = idx >> 4, c = (idx & 15) << 3;
      async_copy_b128(lds_off_u32(&Ks[buf][r * LK + c]), Kb,
                      (unsigned)(((kv0 + r) * E + c) * 2));
    }
  };
  auto loadV = [&](int kv0, v8bf vreg[2]) {
#pragma unroll
    for (int i = 0; i < 2; ++i) {
      int idx = t + i * 256;
      int r = idx >> 4, c = (idx & 15) << 3;
      vreg[i] = *(const v8bf*)&Vb[(size_t)(kv0 + r) * E + c];
    }
  };
  auto scatterV = [&](int buf, const v8bf vreg[2]) {
#pragma unroll
    for (int i = 0; i < 2; ++i) {
      int idx = t + i * 256;
      int r = idx >> 4, c = (idx & 15) << 3;
#pragma unroll
      for (int j = 0; j < 8; ++j) Vt[buf][(c + j) * LV + r] = vreg[i][j];
    }
  };

  // Prologue: tile 0
  stageK(0, 0);
  {
    v8bf v0[2];
    loadV(0, v0);
    scatterV(0, v0);
  }
  wait_async0();
  __syncthreads();

  // Per-wave Q fragments (B operand of S^T = K * Q^T), held all loop long.
  v16bf qb[4];
#pragma unroll
  for (int ds = 0; ds < 4; ++ds) qb[ds] = ldsfrag(&Qs[(wid * 16) * LQ + ds * 32], LQ, lane);

  float l_run = 0.0f;      // per-lane partial (this lane's 16 kv per tile)
  v8f acc[8];              // ctx^T: 128(d) x 16(q) per wave
#pragma unroll
  for (int dt = 0; dt < 8; ++dt) acc[dt] = (v8f)(0.0f);

  const float kpre = (ALPHA * 0.088388347648318447f) / SOFTCAP;  // (1/sqrt128)/50
  const float c2 = 72.134752044448170f;  // SOFTCAP * log2(e); v_exp_f32 is base-2
  const int nkt = S / 32;

  for (int kt = 0; kt < nkt; ++kt) {
    const int cur = kt & 1, nxt = cur ^ 1;
    const bool more = (kt + 1) < nkt;
    v8bf vreg[2];
    if (more) {                       // overlap next tile's traffic with math
      stageK(nxt, (kt + 1) * 32);
      loadV((kt + 1) * 32, vreg);
    }

    // Batch-load all 8 K fragments, then run the 8-WMMA logits chain.
    v16bf kf[2][4];
#pragma unroll
    for (int ds = 0; ds < 4; ++ds) {
      kf[0][ds] = ldsfrag(&Ks[cur][0 * LK + ds * 32], LK, lane);
      kf[1][ds] = ldsfrag(&Ks[cur][16 * LK + ds * 32], LK, lane);
    }
    v8f s0 = (v8f)(0.0f), s1 = (v8f)(0.0f);
#pragma unroll
    for (int ds = 0; ds < 4; ++ds) {
      s0 = wmma_bf16(kf[0][ds], qb[ds], s0);
      s1 = wmma_bf16(kf[1][ds], qb[ds], s1);
    }

    // softcap + exp2 with fixed zero offset (|logits| < 50 -> safe range)
    v16bf pb;
#pragma unroll
    for (int r = 0; r < 8; ++r) {
      float p0 = exp2f(c2 * fast_tanh(s0[r] * kpre));
      float p1 = exp2f(c2 * fast_tanh(s1[r] * kpre));
      l_run += p0 + p1;
      pb[r] = (__bf16)p0;
      pb[8 + r] = (__bf16)p1;
    }

    // Batch-load all 8 V^T fragments, then the 8-WMMA ctx chain.
    v16bf vf[8];
#pragma unroll
    for (int dt = 0; dt < 8; ++dt) vf[dt] = ldsfrag(&Vt[cur][(dt * 16) * LV], LV, lane);
#pragma unroll
    for (int dt = 0; dt < 8; ++dt) acc[dt] = wmma_bf16(vf[dt], pb, acc[dt]);

    if (more) {
      scatterV(nxt, vreg);
      wait_async0();
    }
    __syncthreads();
  }

  // Combine lane halves (lane l and l^16 hold the same query), normalize, store.
  l_run += __shfl_xor(l_run, 16);
  const float inv = 1.0f / l_run;
  const int qn = q0 + wid * 16 + (lane & 15);
  const int rb = (lane >> 4) << 3;
#pragma unroll
  for (int dt = 0; dt < 8; ++dt) {
#pragma unroll
    for (int r = 0; r < 8; ++r) {
      Cb[(size_t)qn * E + dt * 16 + rb + r] = (__bf16)(acc[dt][r] * inv);
    }
  }
}

// ---------------------------------------------------------------------------
// Launch
// ---------------------------------------------------------------------------
extern "C" void kernel_launch(void* const* d_in, const int* in_sizes, int n_in,
                              void* d_out, int out_size, void* d_ws, size_t ws_size,
                              hipStream_t stream) {
  const int B = 2, S = 2048, E = 2048, H = NUM_HEADS;
  const size_t ME = (size_t)B * S * E;  // 8,388,608
  const size_t WE = (size_t)E * E;      // 4,194,304

  // Workspace (bf16): q,k,v | Wq,Wk,Wv,Wo | Qp,Kp,Vp | ctx  (~151 MB)
  __bf16* w = (__bf16*)d_ws;
  __bf16* qb  = w;
  __bf16* kb  = qb + ME;
  __bf16* vb  = kb + ME;
  __bf16* wqb = vb + ME;
  __bf16* wkb = wqb + WE;
  __bf16* wvb = wkb + WE;
  __bf16* wob = wvb + WE;
  __bf16* Qp  = wob + WE;
  __bf16* Kp  = Qp + ME;
  __bf16* Vp  = Kp + ME;
  __bf16* Cx  = Vp + ME;

  __bf16* dsts[7] = {qb, kb, vb, wqb, wkb, wvb, wob};
  size_t  cnts[7] = {ME, ME, ME, WE, WE, WE, WE};
  for (int i = 0; i < 7; ++i) {
    int n4 = (int)(cnts[i] / 4);
    cvt_f32_bf16<<<dim3((n4 + 255) / 256), dim3(256), 0, stream>>>(
        (const float*)d_in[i], dsts[i], (int)cnts[i]);
  }

  dim3 gblk(E / 128, (B * S) / 256);  // (16, 16)
  gemm_xwt_bf16<true><<<gblk, dim3(256), 0, stream>>>(qb, wqb, (void*)Qp, B * S, E, E);
  gemm_xwt_bf16<true><<<gblk, dim3(256), 0, stream>>>(kb, wkb, (void*)Kp, B * S, E, E);
  gemm_xwt_bf16<true><<<gblk, dim3(256), 0, stream>>>(vb, wvb, (void*)Vp, B * S, E, E);

  flash_attn<<<dim3(S / 128, B * H), dim3(256), 0, stream>>>(Qp, Kp, Vp, Cx, B, S, E, H);

  gemm_xwt_bf16<false><<<gblk, dim3(256), 0, stream>>>(Cx, wob, d_out, B * S, E, E);
}
